// EdgeSegNet_2954937499966
// MI455X (gfx1250) — compile-verified, matrix operand order checked by reading, not run on hardware
//
#include <hip/hip_runtime.h>
#include <hip/hip_bf16.h>

// ---------------------------------------------------------------------------
// DGCNN (DynamicEdgeConv x3 + head) for MI455X / gfx1250, wave32 + WMMA.
//
//   prep:   fold BN into W/b, f32->f16, pad K=12->32 / N=13->16, and swizzle
//           weights into WMMA B-fragment-major layout (32B contiguous / lane)
//   knn<C>: brute-force per-cloud kNN (K=16), LDS-tiled, register top-16
//   edge<>: per-point fused edge-MLP: one wave = one point = one 16-row WMMA
//           A tile (16 edges). LDS ping-pong activations, WMMA f16 GEMMs
//           with A hoisted over N tiles, max-over-K via per-lane max+shfl_xor.
//   head:   WMMA GEMM kernels (448->256 relu, 256->128 relu, 128->13)
//
// Input order assumption (setup_inputs insertion order):
//   0:x 1:batch
//   ec1: 2:W0 3:b0 4:g0 5:bt0  6:W1 7:b1 8:g1 9:bt1  10:W2 11:b2
//   ec2: 12:W0 13:b0 14:g0 15:bt0  16:W1 17:b1
//   ec3: 18:W0 19:b0 20:g0 21:bt0  22:W1 23:b1
//   head:24:W0 25:b0 26:W1 27:b1 28:W2 29:b2
// ---------------------------------------------------------------------------

typedef __attribute__((ext_vector_type(16))) _Float16 v16h;
typedef __attribute__((ext_vector_type(8)))  _Float16 v8h;
typedef __attribute__((ext_vector_type(8)))  float    v8f;

#define NPTS   16384   // B*N
#define NPERB  2048
#define KNN    16

// ---------------------------------------------------------------------------
// Fragment loaders.
// A (16-bit, 16x32, ISA 7.12.2): lane (g = lane>>4, m = lane&15) needs K runs
// {k0+g*8 .. +7} and {k0+16+g*8 .. +7}: two contiguous 16-byte loads.
// B: weights pre-swizzled so each lane's 16 values are contiguous (32 bytes).
// ---------------------------------------------------------------------------
__device__ __forceinline__ v16h load_a_lds(const _Float16* L, int ld, int k0, int lane) {
  const int g = (lane >> 4) & 1, m = lane & 15;
  const _Float16* base = L + m * ld + k0 + g * 8;
  v8h lo = *(const v8h*)(base);
  v8h hi = *(const v8h*)(base + 16);
  return __builtin_shufflevector(lo, hi, 0, 1, 2, 3, 4, 5, 6, 7,
                                         8, 9, 10, 11, 12, 13, 14, 15);
}

__device__ __forceinline__ v16h load_a_glb(const _Float16* __restrict__ A, int lda,
                                           int k0, int lane) {
  const int g = (lane >> 4) & 1, m = lane & 15;
  const _Float16* base = A + m * lda + k0 + g * 8;
  v8h lo = *(const v8h*)(base);
  v8h hi = *(const v8h*)(base + 16);
  return __builtin_shufflevector(lo, hi, 0, 1, 2, 3, 4, 5, 6, 7,
                                         8, 9, 10, 11, 12, 13, 14, 15);
}

__device__ __forceinline__ v16h load_b_sw(const _Float16* __restrict__ Wsw, int Ntiles,
                                          int kc, int nt, int lane) {
  return *(const v16h*)(Wsw + (((size_t)(kc * Ntiles + nt) * 32 + lane) << 4));
}

// ---------------------------------------------------------------------------
// Prep: fold BN into weights/bias, f32 -> f16, pad, and swizzle weights into
// B-fragment layout: dst[((kc*Ntiles+nt)*32 + lane)*16 + e] = W'[k(e,lane)][n].
// ---------------------------------------------------------------------------
__global__ void prep_weight_sw(const float* __restrict__ W, const float* __restrict__ gam,
                               _Float16* __restrict__ dst, int Kin, int Nin,
                               int Kpad, int Npad) {
  int t = blockIdx.x * 256 + threadIdx.x;
  if (t >= Kpad * Npad) return;
  const int Ntiles = Npad / 16;
  int e    = t & 15;
  int lane = (t >> 4) & 31;
  int tile = t >> 9;                 // kc * Ntiles + nt
  int nt = tile % Ntiles, kc = tile / Ntiles;
  int g = lane >> 4, n = lane & 15;
  int v = e >> 1, half = e & 1;
  int k   = kc * 32 + ((v & 4) ? 16 : 0) + g * 8 + 2 * (v & 3) + half;
  int col = nt * 16 + n;
  float val = 0.f;
  if (k < Kin && col < Nin) { val = W[k * Nin + col]; if (gam) val *= gam[col]; }
  dst[t] = (_Float16)val;
}

__global__ void prep_bias(const float* __restrict__ b, const float* __restrict__ gam,
                          const float* __restrict__ bt, float* __restrict__ dst,
                          int Nin, int Npad) {
  int t = blockIdx.x * 256 + threadIdx.x;
  if (t >= Npad) return;
  float v = 0.f;
  if (t < Nin) { v = b[t]; if (gam) v = v * gam[t] + bt[t]; }
  dst[t] = v;
}

// ---------------------------------------------------------------------------
// kNN: one thread per query point, 128 queries per block, candidates staged
// in LDS 32 at a time. Top-16 kept in registers with static indexing.
// ---------------------------------------------------------------------------
template <int C>
__global__ __launch_bounds__(128) void knn_kernel(const float* __restrict__ fin,
                                                  int* __restrict__ idx) {
  __shared__ float Lf[32 * C];
  __shared__ float Lsq[32];
  const int tid   = threadIdx.x;
  const int cloud = blockIdx.y;
  const int base  = cloud * NPERB;
  const int q     = base + blockIdx.x * 128 + tid;

  float qf[C];
  float qsq = 0.f;
#pragma unroll
  for (int c = 0; c < C; ++c) { qf[c] = fin[q * C + c]; qsq += qf[c] * qf[c]; }

  float bd[KNN]; int bi[KNN];
#pragma unroll
  for (int s = 0; s < KNN; ++s) { bd[s] = 3.0e38f; bi[s] = base; }
  float bmax = 3.0e38f; int bslot = 0;

  for (int j0 = 0; j0 < NPERB; j0 += 32) {
    __syncthreads();
    for (int t = tid; t < 32 * C; t += 128)
      Lf[t] = fin[(base + j0 + t / C) * C + (t % C)];
    __syncthreads();
    if (tid < 32) {
      float s = 0.f;
#pragma unroll
      for (int c = 0; c < C; ++c) { float v = Lf[tid * C + c]; s += v * v; }
      Lsq[tid] = s;
    }
    __syncthreads();

    for (int cc = 0; cc < 32; ++cc) {
      float dot = 0.f;
#pragma unroll
      for (int c = 0; c < C; ++c) dot = fmaf(qf[c], Lf[cc * C + c], dot);
      float d = qsq + Lsq[cc] - 2.f * dot;
      if (d < bmax) {
        int j = base + j0 + cc;
#pragma unroll
        for (int s = 0; s < KNN; ++s) if (s == bslot) { bd[s] = d; bi[s] = j; }
        bmax = bd[0]; bslot = 0;
#pragma unroll
        for (int s = 1; s < KNN; ++s) if (bd[s] > bmax) { bmax = bd[s]; bslot = s; }
      }
    }
  }
#pragma unroll
  for (int s = 0; s < KNN; ++s) idx[q * KNN + s] = bi[s];
}

// ---------------------------------------------------------------------------
// One MLP layer for one point's 16-edge tile (wave-level), A hoisted over nt.
// Lin: LDS f16 [16][HIN]. If !LAST: relu'd f16 -> Lout. If LAST: max over the
// 16 edge rows (per-lane 8 + xor16 partner) -> outRow (16 channels per tile).
// ---------------------------------------------------------------------------
template <int HIN, int HOUT, bool RELU, bool LAST>
__device__ __forceinline__ void mlp_layer(const _Float16* Lin, _Float16* Lout,
                                          const _Float16* __restrict__ Wsw,
                                          const float* __restrict__ bias,
                                          float* outRow, int lane) {
  constexpr int NT = HOUT / 16;
  const int g = (lane >> 4) & 1, n = lane & 15;

  v8f zero = {};
  v8f acc[NT];
#pragma unroll
  for (int i = 0; i < NT; ++i) acc[i] = zero;

#pragma unroll
  for (int kc = 0; kc < HIN / 32; ++kc) {
    v16h a = load_a_lds(Lin, HIN, kc * 32, lane);
#pragma unroll
    for (int nt = 0; nt < NT; ++nt) {
      v16h b = load_b_sw(Wsw, NT, kc, nt, lane);
      acc[nt] = __builtin_amdgcn_wmma_f32_16x16x32_f16(false, a, false, b,
                                                       (short)0, acc[nt], false, false);
    }
  }

#pragma unroll
  for (int nt = 0; nt < NT; ++nt) {
    float bb = bias[nt * 16 + n];
    if (!LAST) {
#pragma unroll
      for (int r = 0; r < 8; ++r) {
        float v = acc[nt][r] + bb;
        if (RELU) v = fmaxf(v, 0.f);
        Lout[(r + 8 * g) * HOUT + nt * 16 + n] = (_Float16)v;
      }
    } else {
      float m = acc[nt][0];
#pragma unroll
      for (int r = 1; r < 8; ++r) m = fmaxf(m, acc[nt][r]);
      m = fmaxf(m, __shfl_xor(m, 16, 32));   // combine the two lane halves
      if (g == 0) outRow[nt * 16 + n] = m + bb;   // bias commutes with max
    }
  }
}

// ---------------------------------------------------------------------------
// Fused EdgeConv MLP: block = 4 waves, wave w handles point blockIdx.x*4+w.
// ---------------------------------------------------------------------------
template <int CIN, int EIN, int H1, int H2, int H3, int HMAX>
__global__ __launch_bounds__(128) void edge_mlp_kernel(
    const float* __restrict__ fin, const int* __restrict__ idx,
    const _Float16* __restrict__ W1, const float* __restrict__ b1,
    const _Float16* __restrict__ W2, const float* __restrict__ b2,
    const _Float16* __restrict__ W3, const float* __restrict__ b3,
    float* __restrict__ fout) {
  __shared__ __align__(32) _Float16 smem[4][2][16 * HMAX];
  const int wave = threadIdx.x >> 5, lane = threadIdx.x & 31;
  const int p = blockIdx.x * 4 + wave;
  _Float16* bufA = smem[wave][0];
  _Float16* bufB = smem[wave][1];

  // Warm L2/L0 for the weight-stationary loop (global_prefetch_b8).
  __builtin_prefetch(W1 + ((size_t)threadIdx.x << 6), 0, 1);

  // Build the 16 x EIN edge-feature tile [x_i | x_j - x_i | 0-pad].
  for (int t = lane; t < 16 * EIN; t += 32) {
    int row = t / EIN, col = t % EIN;
    float v = 0.f;
    if (col < CIN) {
      v = fin[p * CIN + col];
    } else if (col < 2 * CIN) {
      int j = idx[p * KNN + row];
      int c = col - CIN;
      v = fin[j * CIN + c] - fin[p * CIN + c];
    }
    bufA[t] = (_Float16)v;
  }
  __syncthreads();

  mlp_layer<EIN, H1, true, false>(bufA, bufB, W1, b1, nullptr, lane);
  __syncthreads();
  if constexpr (H3 > 0) {
    mlp_layer<H1, H2, true, false>(bufB, bufA, W2, b2, nullptr, lane);
    __syncthreads();
    mlp_layer<H2, H3, false, true>(bufA, nullptr, W3, b3, fout + p * H3, lane);
  } else {
    mlp_layer<H1, H2, false, true>(bufB, nullptr, W2, b2, fout + p * H2, lane);
  }
}

// ---------------------------------------------------------------------------
// Concat f1|f2|f3 -> f16 feat [NPTS, 448]
// ---------------------------------------------------------------------------
__global__ void concat_kernel(const float* __restrict__ f1, const float* __restrict__ f2,
                              const float* __restrict__ f3, _Float16* __restrict__ feat) {
  int t = blockIdx.x * 256 + threadIdx.x;
  if (t >= NPTS * 448) return;
  int row = t / 448, col = t % 448;
  float v;
  if (col < 64)       v = f1[row * 64 + col];
  else if (col < 192) v = f2[row * 128 + (col - 64)];
  else                v = f3[row * 256 + (col - 192)];
  feat[t] = (_Float16)v;
}

// ---------------------------------------------------------------------------
// WMMA GEMM for the head: out = act(A@W + b).
// A f16 [M,Ka] (row-major), Wsw swizzled [Ka/32][Nb/16][32][16].
// Per-wave 16x16 tile, grid = (Mtiles/4, Nb/16).
// ---------------------------------------------------------------------------
template <bool STORE_F16>
__global__ __launch_bounds__(128) void gemm_kernel(
    const _Float16* __restrict__ A, int Ka,
    const _Float16* __restrict__ Wsw, const float* __restrict__ bias, int Nb,
    void* __restrict__ out, int ldOut, int ncolMax, int relu) {
  const int wave = threadIdx.x >> 5, lane = threadIdx.x & 31;
  const int m0 = (blockIdx.x * 4 + wave) * 16;
  const int nt = blockIdx.y;
  const int Ntiles = Nb / 16;
  const int g = (lane >> 4) & 1, n = lane & 15;
  const _Float16* Arow = A + (size_t)m0 * Ka;

  __builtin_prefetch(Wsw + ((size_t)threadIdx.x << 6), 0, 1);

  v8f acc = {};
  for (int kc = 0; kc < Ka / 32; ++kc) {
    v16h a = load_a_glb(Arow, Ka, kc * 32, lane);
    v16h b = load_b_sw(Wsw, Ntiles, kc, nt, lane);
    acc = __builtin_amdgcn_wmma_f32_16x16x32_f16(false, a, false, b,
                                                 (short)0, acc, false, false);
  }
  int ncol = nt * 16 + n;
  float bb = bias[ncol];
#pragma unroll
  for (int r = 0; r < 8; ++r) {
    float v = acc[r] + bb;
    if (relu) v = fmaxf(v, 0.f);
    int row = m0 + r + 8 * g;
    if (ncol < ncolMax) {
      if (STORE_F16) ((_Float16*)out)[(size_t)row * ldOut + ncol] = (_Float16)v;
      else           ((float*)out)[(size_t)row * ldOut + ncol] = v;
    }
  }
}

// ---------------------------------------------------------------------------
// Host-side orchestration
// ---------------------------------------------------------------------------
extern "C" void kernel_launch(void* const* d_in, const int* in_sizes, int n_in,
                              void* d_out, int out_size, void* d_ws, size_t ws_size,
                              hipStream_t stream) {
  (void)in_sizes; (void)n_in; (void)out_size; (void)ws_size;
  const float* x = (const float*)d_in[0];
  // d_in[1] = batch (implicit: equal-size clouds)

  const float* e1W0 = (const float*)d_in[2];
  const float* e1b0 = (const float*)d_in[3];
  const float* e1g0 = (const float*)d_in[4];
  const float* e1t0 = (const float*)d_in[5];
  const float* e1W1 = (const float*)d_in[6];
  const float* e1b1 = (const float*)d_in[7];
  const float* e1g1 = (const float*)d_in[8];
  const float* e1t1 = (const float*)d_in[9];
  const float* e1W2 = (const float*)d_in[10];
  const float* e1b2 = (const float*)d_in[11];

  const float* e2W0 = (const float*)d_in[12];
  const float* e2b0 = (const float*)d_in[13];
  const float* e2g0 = (const float*)d_in[14];
  const float* e2t0 = (const float*)d_in[15];
  const float* e2W1 = (const float*)d_in[16];
  const float* e2b1 = (const float*)d_in[17];

  const float* e3W0 = (const float*)d_in[18];
  const float* e3b0 = (const float*)d_in[19];
  const float* e3g0 = (const float*)d_in[20];
  const float* e3t0 = (const float*)d_in[21];
  const float* e3W1 = (const float*)d_in[22];
  const float* e3b1 = (const float*)d_in[23];

  const float* hW0 = (const float*)d_in[24];
  const float* hb0 = (const float*)d_in[25];
  const float* hW1 = (const float*)d_in[26];
  const float* hb1 = (const float*)d_in[27];
  const float* hW2 = (const float*)d_in[28];
  const float* hb2 = (const float*)d_in[29];

  // Workspace layout (bytes)
  constexpr size_t OFF_IDX  = 0;                                   // int  [NPTS*16]
  constexpr size_t OFF_F1   = OFF_IDX  + (size_t)NPTS * 16 * 4;
  constexpr size_t OFF_F2   = OFF_F1   + (size_t)NPTS * 64 * 4;
  constexpr size_t OFF_F3   = OFF_F2   + (size_t)NPTS * 128 * 4;
  constexpr size_t OFF_FEAT = OFF_F3   + (size_t)NPTS * 256 * 4;
  constexpr size_t OFF_HH1  = OFF_FEAT + (size_t)NPTS * 448 * 2;
  constexpr size_t OFF_HH2  = OFF_HH1  + (size_t)NPTS * 256 * 2;
  constexpr size_t OFF_WB   = OFF_HH2  + (size_t)NPTS * 128 * 2;

  char* ws = (char*)d_ws;
  int*      idx  = (int*)     (ws + OFF_IDX);
  float*    f1   = (float*)   (ws + OFF_F1);
  float*    f2   = (float*)   (ws + OFF_F2);
  float*    f3   = (float*)   (ws + OFF_F3);
  _Float16* feat = (_Float16*)(ws + OFF_FEAT);
  _Float16* hh1  = (_Float16*)(ws + OFF_HH1);
  _Float16* hh2  = (_Float16*)(ws + OFF_HH2);

  _Float16* Wf = (_Float16*)(ws + OFF_WB);
  size_t o = 0;
  _Float16* We1_0 = Wf + o; o += 32 * 64;     // K padded 12->32
  _Float16* We1_1 = Wf + o; o += 64 * 64;
  _Float16* We1_2 = Wf + o; o += 64 * 64;
  _Float16* We2_0 = Wf + o; o += 128 * 128;
  _Float16* We2_1 = Wf + o; o += 128 * 128;
  _Float16* We3_0 = Wf + o; o += 256 * 256;
  _Float16* We3_1 = Wf + o; o += 256 * 256;
  _Float16* Wh_0  = Wf + o; o += 448 * 256;
  _Float16* Wh_1  = Wf + o; o += 256 * 128;
  _Float16* Wh_2  = Wf + o; o += 128 * 16;    // N padded 13->16
  o = (o + 1) & ~(size_t)1;
  float* Bf = (float*)(Wf + o);
  size_t ob = 0;
  float* Be1_0 = Bf + ob; ob += 64;
  float* Be1_1 = Bf + ob; ob += 64;
  float* Be1_2 = Bf + ob; ob += 64;
  float* Be2_0 = Bf + ob; ob += 128;
  float* Be2_1 = Bf + ob; ob += 128;
  float* Be3_0 = Bf + ob; ob += 256;
  float* Be3_1 = Bf + ob; ob += 256;
  float* Bh_0  = Bf + ob; ob += 256;
  float* Bh_1  = Bf + ob; ob += 128;
  float* Bh_2  = Bf + ob; ob += 16;

  auto pw = [&](const float* W, const float* gm, _Float16* dst,
                int Kin, int Nin, int Kpad, int Npad) {
    int ne = Kpad * Npad;
    prep_weight_sw<<<(ne + 255) / 256, 256, 0, stream>>>(W, gm, dst, Kin, Nin, Kpad, Npad);
  };
  auto pb = [&](const float* b, const float* gm, const float* bt, float* dst,
                int Nin, int Npad) {
    prep_bias<<<(Npad + 255) / 256, 256, 0, stream>>>(b, gm, bt, dst, Nin, Npad);
  };

  // BN folding: W' = W*g (per out-channel), b' = b*g + bt. Last layers plain.
  pw(e1W0, e1g0, We1_0, 12, 64, 32, 64);        pb(e1b0, e1g0, e1t0, Be1_0, 64, 64);
  pw(e1W1, e1g1, We1_1, 64, 64, 64, 64);        pb(e1b1, e1g1, e1t1, Be1_1, 64, 64);
  pw(e1W2, nullptr, We1_2, 64, 64, 64, 64);     pb(e1b2, nullptr, nullptr, Be1_2, 64, 64);
  pw(e2W0, e2g0, We2_0, 128, 128, 128, 128);    pb(e2b0, e2g0, e2t0, Be2_0, 128, 128);
  pw(e2W1, nullptr, We2_1, 128, 128, 128, 128); pb(e2b1, nullptr, nullptr, Be2_1, 128, 128);
  pw(e3W0, e3g0, We3_0, 256, 256, 256, 256);    pb(e3b0, e3g0, e3t0, Be3_0, 256, 256);
  pw(e3W1, nullptr, We3_1, 256, 256, 256, 256); pb(e3b1, nullptr, nullptr, Be3_1, 256, 256);
  pw(hW0, nullptr, Wh_0, 448, 256, 448, 256);   pb(hb0, nullptr, nullptr, Bh_0, 256, 256);
  pw(hW1, nullptr, Wh_1, 256, 128, 256, 128);   pb(hb1, nullptr, nullptr, Bh_1, 128, 128);
  pw(hW2, nullptr, Wh_2, 128, 13, 128, 16);     pb(hb2, nullptr, nullptr, Bh_2, 13, 16);

  const dim3 knnGrid(NPERB / 128, 8);

  // Stage 1: C=6, edge MLP [32(pad12) -> 64 -> 64 -> 64]
  knn_kernel<6><<<knnGrid, 128, 0, stream>>>(x, idx);
  edge_mlp_kernel<6, 32, 64, 64, 64, 64><<<NPTS / 4, 128, 0, stream>>>(
      x, idx, We1_0, Be1_0, We1_1, Be1_1, We1_2, Be1_2, f1);

  // Stage 2: C=64, edge MLP [128 -> 128 -> 128]
  knn_kernel<64><<<knnGrid, 128, 0, stream>>>(f1, idx);
  edge_mlp_kernel<64, 128, 128, 128, 0, 128><<<NPTS / 4, 128, 0, stream>>>(
      f1, idx, We2_0, Be2_0, We2_1, Be2_1, nullptr, nullptr, f2);

  // Stage 3: C=128, edge MLP [256 -> 256 -> 256]
  knn_kernel<128><<<knnGrid, 128, 0, stream>>>(f2, idx);
  edge_mlp_kernel<128, 256, 256, 256, 0, 256><<<NPTS / 4, 128, 0, stream>>>(
      f2, idx, We3_0, Be3_0, We3_1, Be3_1, nullptr, nullptr, f3);

  // Head
  concat_kernel<<<(NPTS * 448) / 256, 256, 0, stream>>>(f1, f2, f3, feat);
  gemm_kernel<true><<<dim3(NPTS / 64, 16), 128, 0, stream>>>(
      feat, 448, Wh_0, Bh_0, 256, hh1, 256, 256, 1);
  gemm_kernel<true><<<dim3(NPTS / 64, 8), 128, 0, stream>>>(
      hh1, 256, Wh_1, Bh_1, 128, hh2, 128, 128, 1);
  gemm_kernel<false><<<dim3(NPTS / 64, 1), 128, 0, stream>>>(
      hh2, 128, Wh_2, Bh_2, 16, d_out, 13, 13, 0);
}
